// Bottleneck_40553081209079
// MI455X (gfx1250) — matrix operand order repaired and checked by reading, not compiled, not run on hardware
//
#include <hip/hip_runtime.h>
#include <stdint.h>

typedef __attribute__((ext_vector_type(16))) __bf16       v16bf;
typedef __attribute__((ext_vector_type(8)))  float        v8f;
typedef __attribute__((ext_vector_type(4)))  unsigned int u32x4;
typedef __attribute__((ext_vector_type(2)))  unsigned int u32x2;

#define NB    16
#define C_IN  16
#define C_OUT 64
#define HH    256
#define WW    256
#define KTOT  144          // C_IN * 9
#define PXWG  128          // pixels per workgroup (along W)
#define XCOLS 130          // bf16 staged cols (halo of 1 each side)
#define XFRUN 136          // fp32 staged cols per (row,ci) run (16B-aligned window w0-4..w0+131)
#define XF_N128 (3 * C_IN * (XFRUN / 4))   // 1632 async b128 ops for x
#define WS_BYTES (C_OUT * KTOT * 2)        // 18,432 B per bf16 weight plane
#define WS_N128  ((2 * WS_BYTES) / 16)     // 2304 async b128 ops for both planes

// LDS layout (byte offsets). fp32 x staging aliases the x bf16 planes
// (4B fp32 == 2B hi + 2B lo), reused in-place after a barrier.
#define XS_HI_OFF 0
#define XS_LO_OFF 12480            // 3*130*16*2
#define XF_OFF    0                // fp32 x patch [3][16][136]  (26,112 B)
#define REGA_SZ   26112
#define GAP_OFF   24960            // stale fp32 tail [24,960, 26,112): zeroed (NaN hygiene)
#define WS_HI_OFF REGA_SZ          // prepped bf16 weights [co][144], hi plane
#define WS_LO_OFF (REGA_SZ + WS_BYTES)
#define SMEM_U4   ((REGA_SZ + 2 * WS_BYTES) / 16)   // 62,976 B total

union Frag  { u32x4 q[2]; v16bf v; };
union Pack4 { __bf16 b[4]; u32x2 d; };

__device__ __forceinline__ void async_copy_b128(uint32_t lds_byte_addr, const void* gptr) {
  // GV mode: LDS[vdst] = MEM[vaddr64], tracked by ASYNCcnt, no VGPR round-trip
  asm volatile("global_load_async_to_lds_b128 %0, %1, off"
               :: "v"(lds_byte_addr), "v"((uint64_t)(uintptr_t)gptr)
               : "memory");
}

// ---- one-block prep: OIHW fp32 weights -> [co][k] split-bf16 planes in d_ws
__global__ __launch_bounds__(256)
void prep_weights(const float* __restrict__ Wg, __bf16* __restrict__ wsp) {
  __bf16* wshi = wsp;
  __bf16* wslo = wsp + C_OUT * KTOT;
  const int t  = threadIdx.x;
  const int co = t >> 2;             // 0..63
  const int kb = (t & 3) * 36;       // k block within 144
  #pragma unroll
  for (int i = 0; i < 36; ++i) {
    int k   = kb + i;
    int tap = k >> 4, ci = k & 15;
    float f  = Wg[(co * C_IN + ci) * 9 + tap];
    __bf16 hi = (__bf16)f;
    __bf16 lo = (__bf16)(f - (float)hi);
    wshi[co * KTOT + k] = hi;
    wslo[co * KTOT + k] = lo;
  }
}

__global__ __launch_bounds__(256)
void conv3x3_implicit_gemm_wmma(const float* __restrict__ x,
                                const __bf16* __restrict__ wsp,
                                float* __restrict__ out) {
  __shared__ u32x4 smem4[SMEM_U4];
  char* smem = (char*)smem4;
  const uint32_t lds_base = (uint32_t)(uintptr_t)(void*)smem;

  const int tid  = threadIdx.x;
  const int bid  = blockIdx.x;
  const int wblk = bid & 1;                // W / PXWG = 2
  const int h    = (bid >> 1) & (HH - 1);
  const int n    = bid >> 9;
  const int w0   = wblk * PXWG;

  // ================= Phase A: async global -> LDS =================
  // x patch (raw fp32): rows h-1..h+1, cols w0-4..w0+131 (aligned), 16 ci
  for (int idx = tid; idx < XF_N128; idx += 256) {
    int j   = idx % (XFRUN / 4);
    int rc  = idx / (XFRUN / 4);
    int ci  = rc & (C_IN - 1);
    int row = rc >> 4;
    int gh   = min(max(h + row - 1, 0), HH - 1);        // clamp; OOB zeroed later
    int gcol = min(max(w0 - 4 + 4 * j, 0), WW - 4);     // clamp; OOB zeroed later
    const float* gp = x + (((n * C_IN + ci) * HH + gh) * WW + gcol);
    async_copy_b128(lds_base + XF_OFF + (uint32_t)(((row * C_IN + ci) * XFRUN + 4 * j) * 4), gp);
  }
  // prepped split-bf16 weights: straight contiguous copy, ready to use
  for (int idx = tid; idx < WS_N128; idx += 256) {
    async_copy_b128(lds_base + WS_HI_OFF + (uint32_t)(idx * 16),
                    (const char*)wsp + idx * 16);
  }
  asm volatile("s_wait_asynccnt 0" ::: "memory");
  __syncthreads();

  // ====== Phase B: in-place x fp32 -> split-bf16 (shift-only indexing) ======
  // thread t: ci group (t&3)*4, cols (t>>2)+{0,64,128}, all 3 rows; 4-el groups.
  const float* Xf = (const float*)(smem + XF_OFF);     // [3][16][136]
  const int ci4 = (tid & 3) * 4;
  const int c0  = tid >> 2;
  float fr[3][3][4];
  #pragma unroll
  for (int row = 0; row < 3; ++row) {
    const int gh_ok = ((unsigned)(h + row - 1) < (unsigned)HH) ? 1 : 0;
    #pragma unroll
    for (int i = 0; i < 3; ++i) {
      const int col = c0 + 64 * i;
      if (i < 2 || c0 < 2) {                           // col < 130
        const int gw_ok = ((unsigned)(w0 + col - 1) < (unsigned)WW) ? 1 : 0;
        const float* rp = Xf + (row * C_IN + ci4) * XFRUN + col + 3;
        #pragma unroll
        for (int j = 0; j < 4; ++j)
          fr[row][i][j] = (gh_ok & gw_ok) ? rp[j * XFRUN] : 0.0f;
      }
    }
  }
  __syncthreads();
  __bf16* xs_hi = (__bf16*)(smem + XS_HI_OFF);         // [3][130][16] flat
  __bf16* xs_lo = (__bf16*)(smem + XS_LO_OFF);
  #pragma unroll
  for (int row = 0; row < 3; ++row) {
    #pragma unroll
    for (int i = 0; i < 3; ++i) {
      const int col = c0 + 64 * i;
      if (i < 2 || c0 < 2) {
        Pack4 ph, pl;
        #pragma unroll
        for (int j = 0; j < 4; ++j) {
          float f = fr[row][i][j];
          __bf16 hi = (__bf16)f;
          ph.b[j] = hi;
          pl.b[j] = (__bf16)(f - (float)hi);
        }
        const int e = (row * XCOLS + col) * C_IN + ci4;   // 8B-aligned (ci4 mult of 4)
        *(u32x2*)(xs_hi + e) = ph.d;
        *(u32x2*)(xs_lo + e) = pl.d;
      }
    }
  }
  // NaN hygiene: the pad-lane (tap==9) B reads may alias this stale fp32 tail;
  // products are killed by zeroed A columns, but operands must stay finite.
  if (tid < 144) {
    const u32x2 z2 = {0u, 0u};
    *(u32x2*)(smem + GAP_OFF + tid * 8) = z2;
  }
  __syncthreads();

  // ======================= Phase C: WMMA main loop ==========================
  const int wave = tid >> 5;
  const int lane = tid & 31;
  const int g    = lane >> 4;      // lane half
  const int ln   = lane & 15;
  const int ct   = wave & 3;       // c_out tile (16 channels)
  const int pxg  = wave >> 2;      // pixel group (64 pixels)
  const int coA  = ct * 16 + ln;   // A-matrix row held by this lane
  const __bf16* ws_hi = (const __bf16*)(smem + WS_HI_OFF);
  const __bf16* ws_lo = (const __bf16*)(smem + WS_LO_OFF);

  const v8f   zacc = {0.f, 0.f, 0.f, 0.f, 0.f, 0.f, 0.f, 0.f};
  const u32x4 zq   = {0u, 0u, 0u, 0u};

  v8f acc[4];
  #pragma unroll
  for (int pt = 0; pt < 4; ++pt) acc[pt] = zacc;

  #pragma unroll
  for (int c = 0; c < 5; ++c) {    // K = 144 padded to 5 chunks of 32
    Frag a_hi, a_lo;
    {
      const int k0 = 32 * c + 16 * g;
      if (c < 4 || g == 0) {
        a_hi.q[0] = *(const u32x4*)(ws_hi + coA * KTOT + k0);
        a_lo.q[0] = *(const u32x4*)(ws_lo + coA * KTOT + k0);
        a_hi.q[1] = *(const u32x4*)(ws_hi + coA * KTOT + k0 + 8);
        a_lo.q[1] = *(const u32x4*)(ws_lo + coA * KTOT + k0 + 8);
      } else {                     // k >= 144 pad: zero A columns kill the products
        a_hi.q[0] = zq;
        a_lo.q[0] = zq;
        a_hi.q[1] = zq;
        a_lo.q[1] = zq;
      }
    }
    // B fragment: lane holds N=ln, K = 16*g + 2v+h -> single tap per lane-half,
    // 16 contiguous ci in LDS => two b128 loads. tap==9 (c==4, g==1) reads
    // harmless finite in-bounds garbage that is multiplied by zero A.
    const int tapB = 2 * c + g;
    const int rowB = tapB / 3;
    const int kwB  = tapB - rowB * 3;

    #pragma unroll
    for (int pt = 0; pt < 4; ++pt) {
      const int px = pxg * 64 + pt * 16 + ln;
      const __bf16* bp_hi = xs_hi + (rowB * XCOLS + px + kwB) * C_IN;
      const __bf16* bp_lo = xs_lo + (rowB * XCOLS + px + kwB) * C_IN;
      Frag b_hi, b_lo;
      b_hi.q[0] = *(const u32x4*)(bp_hi);
      b_hi.q[1] = *(const u32x4*)(bp_hi + 8);
      b_lo.q[0] = *(const u32x4*)(bp_lo);
      b_lo.q[1] = *(const u32x4*)(bp_lo + 8);
      acc[pt] = __builtin_amdgcn_wmma_f32_16x16x32_bf16(
          false, a_hi.v, false, b_hi.v, (short)0, acc[pt], false, false);
      acc[pt] = __builtin_amdgcn_wmma_f32_16x16x32_bf16(
          false, a_hi.v, false, b_lo.v, (short)0, acc[pt], false, false);
      acc[pt] = __builtin_amdgcn_wmma_f32_16x16x32_bf16(
          false, a_lo.v, false, b_hi.v, (short)0, acc[pt], false, false);
    }
  }

  // ================== Phase D: non-temporal output stores ===================
  // Streaming 268 MB; NT keeps the 192 MB L2 free for the reused input.
  #pragma unroll
  for (int pt = 0; pt < 4; ++pt) {
    const int wgl = w0 + pxg * 64 + pt * 16 + ln;
    #pragma unroll
    for (int v = 0; v < 8; ++v) {
      const int co = ct * 16 + 8 * g + v;
      __builtin_nontemporal_store(acc[pt][v],
          out + ((n * C_OUT + co) * HH + h) * WW + wgl);
    }
  }
}

extern "C" void kernel_launch(void* const* d_in, const int* in_sizes, int n_in,
                              void* d_out, int out_size, void* d_ws, size_t ws_size,
                              hipStream_t stream) {
  const float* x  = (const float*)d_in[0];
  const float* Wp = (const float*)d_in[1];
  float* out      = (float*)d_out;
  __bf16* wsp     = (__bf16*)d_ws;    // 36,864 B of scratch for split weights

  prep_weights<<<dim3(1), dim3(256), 0, stream>>>(Wp, wsp);
  dim3 grid(NB * HH * (WW / PXWG));   // 16 * 256 * 2 = 8192 workgroups
  dim3 block(256);                    // 8 wave32s
  conv3x3_implicit_gemm_wmma<<<grid, block, 0, stream>>>(x, wsp, out);
}
